// GraphAttention_23699629539642
// MI455X (gfx1250) — matrix operand order repaired and checked by reading, not compiled, not run on hardware
//
#include <hip/hip_runtime.h>

// Problem constants (match reference)
#define BB       4
#define NN       2048
#define IN_FEAT  128
#define OUT_F    64
#define HEADS    4
#define NEG_MASK (-1e16f)

typedef __attribute__((ext_vector_type(2)))  float  v2f;
typedef __attribute__((ext_vector_type(8)))  float  v8f;
typedef __attribute__((ext_vector_type(16))) __bf16 v16bf;

// -------------------------------------------------------------------------
// Kernel 1: fs = X @ W_self, fn = X @ W_neigh  per (b,h), f32 WMMA 16x16x4.
// One wave per 16x16 output tile; shared A fragment, two B fragments.
// wave id bits: [1:0]=o_tile(4) [8:2]=m_tile(128) [10:9]=h [12:11]=b
// -------------------------------------------------------------------------
__global__ __launch_bounds__(128) void gat_proj(
    const float* __restrict__ X, const float* __restrict__ Ws,
    const float* __restrict__ Wn, float* __restrict__ fs, float* __restrict__ fn)
{
    const int wave = blockIdx.x * 4 + (threadIdx.x >> 5);
    const int lane = threadIdx.x & 31;
    const int o_t = wave & 3;
    const int m_t = (wave >> 2) & 127;
    const int h   = (wave >> 9) & 3;
    const int b   = wave >> 11;

    const int m     = lane & 15;       // row within tile (A) / col (B,C)
    const int khalf = (lane >> 4) * 2; // K offset split across lane halves

    const float* xrow = X + ((size_t)b * NN + (size_t)m_t * 16 + m) * IN_FEAT;
    const float* wsp  = Ws + (size_t)h * IN_FEAT * OUT_F + o_t * 16 + m;
    const float* wnp  = Wn + (size_t)h * IN_FEAT * OUT_F + o_t * 16 + m;

    v8f cS = {}; v8f cN = {};
#pragma unroll 4
    for (int k = 0; k < IN_FEAT; k += 4) {
        const int kr = k + khalf;
        v2f a = *(const v2f*)(xrow + kr);                       // A: 16x4 f32
        v2f bs; bs.x = wsp[(size_t)kr * OUT_F]; bs.y = wsp[(size_t)(kr + 1) * OUT_F];
        v2f bn; bn.x = wnp[(size_t)kr * OUT_F]; bn.y = wnp[(size_t)(kr + 1) * OUT_F];
        cS = __builtin_amdgcn_wmma_f32_16x16x4_f32(false, a, false, bs, (short)0, cS, false, false);
        cN = __builtin_amdgcn_wmma_f32_16x16x4_f32(false, a, false, bn, (short)0, cN, false, false);
    }

    const size_t rowbase = ((size_t)b * HEADS + h) * NN + (size_t)m_t * 16;
    const int ncol = lane & 15;
    const int mofs = (lane >> 4) * 8;
#pragma unroll
    for (int v = 0; v < 8; ++v) {
        size_t idx = (rowbase + v + mofs) * OUT_F + o_t * 16 + ncol;
        fs[idx] = cS[v];
        fn[idx] = cN[v];
    }
}

// -------------------------------------------------------------------------
// Kernel 2: att_s[bhn] = fs[bhn,:]·a_self[h], att_n likewise. One thread/node.
// -------------------------------------------------------------------------
__global__ __launch_bounds__(256) void gat_logits(
    const float* __restrict__ fs, const float* __restrict__ fn,
    const float* __restrict__ a_s, const float* __restrict__ a_n,
    float* __restrict__ att_s, float* __restrict__ att_n)
{
    const int t = blockIdx.x * blockDim.x + threadIdx.x;
    if (t >= BB * HEADS * NN) return;
    const int h = (t / NN) % HEADS;
    const float4* fr = (const float4*)(fs + (size_t)t * OUT_F);
    const float4* gr = (const float4*)(fn + (size_t)t * OUT_F);
    const float4* as = (const float4*)(a_s + (size_t)h * OUT_F);
    const float4* an = (const float4*)(a_n + (size_t)h * OUT_F);
    float s = 0.f, n = 0.f;
#pragma unroll
    for (int q = 0; q < OUT_F / 4; ++q) {
        float4 f = fr[q], a = as[q];
        s += f.x * a.x + f.y * a.y + f.z * a.z + f.w * a.w;
        float4 g = gr[q], c = an[q];
        n += g.x * c.x + g.y * c.y + g.z * c.z + g.w * c.w;
    }
    att_s[t] = s;
    att_n[t] = n;
}

// -------------------------------------------------------------------------
// Kernel 3: one workgroup per attention row (b,h,i). Fused lrelu/mask/
// sigmoid kept in registers, block-reduced sum -> beta, scaled row written
// once into the output att region. Also stash beta per node.
// -------------------------------------------------------------------------
__global__ __launch_bounds__(256) void gat_att_row(
    const float* __restrict__ A, const float* __restrict__ att_s,
    const float* __restrict__ att_n, float* __restrict__ att_out,
    float* __restrict__ beta_out)
{
    const int row = blockIdx.x;       // ((b*H+h)*N + i)
    const int i   = row % NN;
    const int bh  = row / NN;
    const int b   = bh / HEADS;
    const int t   = threadIdx.x;

    const float s_i = att_s[row];
    const float* arow = A + ((size_t)b * NN + i) * NN;
    const float* anv  = att_n + (size_t)bh * NN;

    float vals[NN / 256];
    float lsum = 0.f;
#pragma unroll
    for (int r = 0; r < NN / 256; ++r) {
        const int j = t + r * 256;
        float x = s_i + anv[j];
        x = (x > 0.f) ? x : 0.2f * x;            // leaky_relu(0.2)
        x += NEG_MASK * (1.0f - arow[j]);        // mask non-edges
        float sg = 1.0f / (1.0f + __expf(-x));   // sigmoid (masked -> 0)
        vals[r] = sg;
        lsum += sg;
    }
    __shared__ float red[256];
    red[t] = lsum;
    __syncthreads();
    for (int ofs = 128; ofs > 0; ofs >>= 1) {
        if (t < ofs) red[t] += red[t + ofs];
        __syncthreads();
    }
    const float beta = 1.0f / (1.0f + red[0]);

    float* orow = att_out + (size_t)row * NN;
#pragma unroll
    for (int r = 0; r < NN / 256; ++r)
        orow[t + r * 256] = vals[r] * beta;
    if (t == 0) beta_out[row] = beta;
}

// -------------------------------------------------------------------------
// Kernel 4: node_feat = beta*fs + att @ fn, ELU, head-concat store.
// bf16 WMMA 16x16x32, f32 accumulate. Block = 4 waves; wave w owns o-tile w.
// LDS holds bf16 tiles (converted once at staging); fragment gathers are
// packed 32-bit LDS loads. Global loads of tile k+1 are issued between the
// store barrier and the WMMA consume of tile k (register pipelining), and
// the att stream is prefetched two tiles ahead (global_prefetch_b8).
// blk bits: [6:0]=m_tile(128), [10:7]=bh(16)
// -------------------------------------------------------------------------
#define B_STRIDE 34   // bf16 elements per transposed-fn row (pad vs 32)

__global__ __launch_bounds__(128) void gat_aggregate(
    const float* __restrict__ att, const float* __restrict__ fn,
    const float* __restrict__ fs, const float* __restrict__ beta,
    float* __restrict__ out)
{
    const int blk = blockIdx.x;
    const int m_t = blk & 127;
    const int bh  = blk >> 7;
    const int h   = bh & (HEADS - 1);
    const int b   = bh / HEADS;
    const int tid  = threadIdx.x;
    const int wv   = tid >> 5;
    const int lane = tid & 31;
    const int o0   = wv * 16;

    __shared__ __align__(8) __bf16 ldsA[16 * 32];           // att tile [m][k]
    __shared__ __align__(8) __bf16 ldsB[64 * B_STRIDE];     // fn tile [o][k]

    const float* attbase = att + ((size_t)bh * NN + (size_t)m_t * 16) * NN;
    const float* fnbase  = fn + (size_t)bh * NN * OUT_F;

    const int m  = lane & 15;
    const int n  = lane & 15;
    const int hi = lane >> 4;              // lane-half selector

    // register staging for the pipelined tile
    float ra[4];    // 4 att elements / thread  (16x32 tile, 128 threads)
    float rb[16];   // 16 fn elements / thread  (32x64 tile)

    // preload tile 0
#pragma unroll
    for (int r = 0; r < 4; ++r) {
        int e = tid + r * 128;
        ra[r] = attbase[(size_t)(e >> 5) * NN + (e & 31)];
    }
#pragma unroll
    for (int r = 0; r < 16; ++r) {
        int e = tid + r * 128;
        rb[r] = fnbase[(size_t)(e >> 6) * OUT_F + (e & 63)];
    }

    v8f acc = {};
    for (int k0 = 0; k0 < NN; k0 += 32) {
        __syncthreads();                         // prior consume finished
        // store staged tile to LDS, converting f32 -> bf16 once
#pragma unroll
        for (int r = 0; r < 4; ++r)
            ldsA[tid + r * 128] = (__bf16)ra[r];
#pragma unroll
        for (int r = 0; r < 16; ++r) {
            int e = tid + r * 128;
            ldsB[(e & 63) * B_STRIDE + (e >> 6)] = (__bf16)rb[r];
        }
        __syncthreads();                         // tile visible

        const int kn = k0 + 32;
        if (kn < NN) {
            // prefetch two tiles ahead on the streaming att tensor
            const int kp = k0 + 64;
            if (kp < NN)
                __builtin_prefetch(attbase + (size_t)(tid >> 5) * NN + kp + (tid & 31), 0, 3);
            // pipelined global loads of tile k+1 (overlap WMMA below)
#pragma unroll
            for (int r = 0; r < 4; ++r) {
                int e = tid + r * 128;
                ra[r] = attbase[(size_t)(e >> 5) * NN + kn + (e & 31)];
            }
#pragma unroll
            for (int r = 0; r < 16; ++r) {
                int e = tid + r * 128;
                rb[r] = fnbase[(size_t)(kn + (e >> 6)) * OUT_F + (e & 63)];
            }
        }

        // A fragment: 16-bit A 16x32 layout (ISA 7.12.2); packed 2xbf16 loads
        union { v16bf v; unsigned int u[8]; } af, bf;
#pragma unroll
        for (int p = 0; p < 8; ++p) {
            int kb = ((p < 4) ? 0 : 16) + hi * 8 + 2 * (p & 3);
            af.u[p] = *(const unsigned int*)&ldsA[m * 32 + kb];
        }
        // B fragment: same K order as A (16-bit 16x32 layout)
        {
            const __bf16* bp = &ldsB[(o0 + n) * B_STRIDE + hi * 8];
#pragma unroll
            for (int q = 0; q < 8; ++q)
                bf.u[q] = *(const unsigned int*)(bp + 2 * q + ((q < 4) ? 0 : 8));
        }

        acc = __builtin_amdgcn_wmma_f32_16x16x32_bf16(false, af.v, false, bf.v,
                                                      (short)0, acc, false, false);
    }

    // epilogue: + beta*fs, ELU, store [B,N,H*O]
    const int mofs = (lane >> 4) * 8;
#pragma unroll
    for (int v = 0; v < 8; ++v) {
        const int mrow = m_t * 16 + v + mofs;
        const size_t node = (size_t)bh * NN + mrow;
        const float bta = beta[node];
        float val = acc[v] + bta * fs[node * OUT_F + o0 + n];
        val = (val > 0.f) ? val : (__expf(val) - 1.0f);   // ELU
        out[((size_t)b * NN + mrow) * (HEADS * OUT_F) + h * OUT_F + o0 + n] = val;
    }
}

// -------------------------------------------------------------------------
extern "C" void kernel_launch(void* const* d_in, const int* in_sizes, int n_in,
                              void* d_out, int out_size, void* d_ws, size_t ws_size,
                              hipStream_t stream)
{
    (void)in_sizes; (void)n_in; (void)out_size; (void)ws_size;
    const float* X   = (const float*)d_in[0];
    const float* A   = (const float*)d_in[1];
    const float* Ws  = (const float*)d_in[2];
    const float* Wn  = (const float*)d_in[3];
    const float* a_s = (const float*)d_in[4];
    const float* a_n = (const float*)d_in[5];

    // Workspace layout (floats)
    const size_t FS_N = (size_t)BB * HEADS * NN * OUT_F;   // 2,097,152
    const size_t SC_N = (size_t)BB * HEADS * NN;           // 32,768
    float* ws_fs   = (float*)d_ws;
    float* ws_fn   = ws_fs + FS_N;
    float* ws_atts = ws_fn + FS_N;
    float* ws_attn = ws_atts + SC_N;
    float* ws_beta = ws_attn + SC_N;

    // Output: [out (B*N*H*O)] then [att (B*H*N*N)]
    float* out0    = (float*)d_out;
    float* att_out = out0 + (size_t)BB * NN * HEADS * OUT_F;

    // 1. projections (f32 WMMA)
    {
        const int waves = BB * HEADS * (NN / 16) * (OUT_F / 16); // 8192
        gat_proj<<<waves / 4, 128, 0, stream>>>(X, Ws, Wn, ws_fs, ws_fn);
    }
    // 2. attention logits
    gat_logits<<<(BB * HEADS * NN + 255) / 256, 256, 0, stream>>>(
        ws_fs, ws_fn, a_s, a_n, ws_atts, ws_attn);
    // 3. fused sigmoid-attention row pass (writes scaled att to output)
    gat_att_row<<<BB * HEADS * NN, 256, 0, stream>>>(
        A, ws_atts, ws_attn, att_out, ws_beta);
    // 4. aggregation (bf16 WMMA) + ELU + concat
    gat_aggregate<<<BB * HEADS * (NN / 16), 128, 0, stream>>>(
        att_out, ws_fn, ws_fs, ws_beta, out0);
}
